// GKT_23046794510941
// MI455X (gfx1250) — matrix operand order repaired and checked by reading, not compiled, not run on hardware
//
#include <hip/hip_runtime.h>
#include <hip/hip_bf16.h>
#include <math.h>

// ---------------------------------------------------------------------------
// GKT forward for MI455X (gfx1250, wave32, WMMA 16x16x32 f16 + TDM).
// ---------------------------------------------------------------------------
#define B_ 64
#define C_ 2048
#define H_ 64
#define E_ 64
#define K_ 2
#define Q_ 4096
#define D_ 128
#define NT_ (B_ * C_ / 16)   // 8192 row-tiles of 16 rows

typedef __attribute__((ext_vector_type(16))) _Float16 v16h;
typedef __attribute__((ext_vector_type(8)))  float    v8f;
typedef __attribute__((ext_vector_type(4)))  unsigned int u32x4;
typedef __attribute__((ext_vector_type(8)))  int i32x8;
typedef __attribute__((ext_vector_type(4)))  int i32x4;

#if defined(__has_builtin)
#if __has_builtin(__builtin_amdgcn_tensor_load_to_lds)
#define USE_TDM 1
#endif
#endif
#ifndef USE_TDM
#define USE_TDM 0
#endif

__device__ __forceinline__ float sigf(float x) { return 1.0f / (1.0f + __expf(-x)); }

__device__ __forceinline__ v8f wmma_f16(v16h a, v16h b, v8f c) {
  return __builtin_amdgcn_wmma_f32_16x16x32_f16(false, a, false, b, (short)0, c,
                                                false, false);
}

// ---- pre-packed B fragments (f16, per-lane WMMA layout) -------------------
// frag f, lane l: 16 halves at fb + f*512 + l*16  (32B/lane, contiguous)
#define FRAG_W1(k, j, kk)  (((k) << 3) + ((j) << 1) + (kk))        // 0..15
#define FRAG_W2(k, j, kk)  (16 + ((k) << 3) + ((j) << 1) + (kk))   // 16..31
#define FRAG_EW(j, kk)     (32 + ((j) << 1) + (kk))                // 32..39
#define FRAG_AW(j, kk)     (40 + ((j) << 1) + (kk))                // 40..47
#define FRAG_WIH(jt, kk)   (48 + ((jt) << 1) + (kk))               // 48..71
#define FRAG_WHH(jt, kk)   (72 + ((jt) << 1) + (kk))               // 72..95
#define NFRAG 96

__device__ __forceinline__ v16h bfrag_ld(const _Float16* __restrict__ fb, int f,
                                         int lane) {
  return *(const v16h*)(fb + ((f << 9) + (lane << 4)));
}

// B-frag element: B[kb+e][n], n = lane%16 + 16*jt, kb = 32*kk + (lane>=16?16:0)
__global__ __launch_bounds__(32) void k_packfrags(
    const float* __restrict__ fnW1, const float* __restrict__ fnW2,
    const float* __restrict__ eW, const float* __restrict__ aW,
    const float* __restrict__ wih, const float* __restrict__ whh,
    _Float16* __restrict__ fb) {
  int f = blockIdx.x;
  int lane = threadIdx.x;
  int n15 = lane & 15;
  int lup = (lane & 16) ? 16 : 0;
  v16h r;
  if (f < 16) {                       // fn_W1[k][128:192] (64x64), direct
    int k = f >> 3, j = (f >> 1) & 3, kk = f & 1;
    const float* W = fnW1 + k * 256 * 64 + 128 * 64;
    int n = n15 + (j << 4), kb = (kk << 5) + lup;
#pragma unroll
    for (int e = 0; e < 16; ++e) r[e] = (_Float16)W[(kb + e) * 64 + n];
  } else if (f < 32) {                // fn_W2[k] (64x64), direct
    int g = f - 16, k = g >> 3, j = (g >> 1) & 3, kk = g & 1;
    const float* W = fnW2 + k * 64 * 64;
    int n = n15 + (j << 4), kb = (kk << 5) + lup;
#pragma unroll
    for (int e = 0; e < 16; ++e) r[e] = (_Float16)W[(kb + e) * 64 + n];
  } else if (f < 40) {                // eag_eW (64x64), direct
    int g = f - 32, j = g >> 1, kk = g & 1;
    int n = n15 + (j << 4), kb = (kk << 5) + lup;
#pragma unroll
    for (int e = 0; e < 16; ++e) r[e] = (_Float16)eW[(kb + e) * 64 + n];
  } else if (f < 48) {                // eag_aW (64x64), direct
    int g = f - 40, j = g >> 1, kk = g & 1;
    int n = n15 + (j << 4), kb = (kk << 5) + lup;
#pragma unroll
    for (int e = 0; e < 16; ++e) r[e] = (_Float16)aW[(kb + e) * 64 + n];
  } else if (f < 72) {                // gru_wih.T : B[k][n] = wih[n*64+k]
    int g = f - 48, jt = g >> 1, kk = g & 1;
    int n = n15 + (jt << 4), kb = (kk << 5) + lup;
#pragma unroll
    for (int e = 0; e < 16; ++e) r[e] = (_Float16)wih[n * 64 + kb + e];
  } else {                            // gru_whh.T
    int g = f - 72, jt = g >> 1, kk = g & 1;
    int n = n15 + (jt << 4), kb = (kk << 5) + lup;
#pragma unroll
    for (int e = 0; e < 16; ++e) r[e] = (_Float16)whh[n * 64 + kb + e];
  }
  *(v16h*)(fb + ((f << 9) + (lane << 4))) = r;
}

// A fragment (16x32 slice, K-half kk) from an LDS f32 tile (16 rows x 64).
__device__ __forceinline__ v16h afrag_lds(const float* t, int kk, int lane) {
  asm volatile("s_wait_dscnt 0" ::: "memory");
  int m = lane & 15;
  int kb = (kk << 5) + ((lane & 16) ? 16 : 0);
  const float* row = t + m * 64 + kb;
  v16h r;
#pragma unroll
  for (int e = 0; e < 16; ++e) r[e] = (_Float16)row[e];
  return r;
}

// ---- Tensor Data Mover: DMA one 16x64 f32 tile (4 KB) global -> LDS -------
#if USE_TDM
__device__ __forceinline__ void tdm_load_tile(const float* gsrc, void* ldsdst) {
  unsigned long long ga = (unsigned long long)(size_t)gsrc;
  u32x4 g0;
  g0[0] = 1u;                                        // count=1, user descriptor
  g0[1] = (unsigned int)(size_t)ldsdst;              // lds_addr (LDS aperture low bits)
  g0[2] = (unsigned int)ga;                          // global_addr[31:0]
  g0[3] = ((unsigned int)(ga >> 32) & 0x1FFFFFFu) | (2u << 30);  // addr[56:32] | type=2
  i32x8 g1;
  g1[0] = 0x00020000;          // workgroup_mask=0 (not in cluster), data_size=4B
  g1[1] = (int)(1024u << 16);  // tensor_dim0 = 1024 elements
  g1[2] = (int)(1u << 16);     // tensor_dim1 = 1
  g1[3] = (int)(1024u << 16);  // tile_dim0 = 1024
  g1[4] = 1;                   // tile_dim1 = 1, tile_dim2 = 0
  g1[5] = 1024;                // tensor_dim0_stride = 1024
  g1[6] = 0;
  g1[7] = 0;
  i32x4 z4 = {0, 0, 0, 0};
#if __has_include(<hip/amd_detail/amd_gfx1250_TDM.h>)
  i32x8 z8 = {0, 0, 0, 0, 0, 0, 0, 0};
  __builtin_amdgcn_tensor_load_to_lds(g0, g1, z4, z4, z8, 0);
#else
  __builtin_amdgcn_tensor_load_to_lds(g0, g1, z4, z4, 0);
#endif
}
#endif

// h2_k tile (post-relu, pre-norm) for one 16-row tile of the fn MLP.
__device__ __forceinline__ void compute_h2(
    int k, int b, int c0, int qtb, v16h a0, v16h a1,
    const _Float16* __restrict__ fb, const float* __restrict__ fnb2,
    const float* __restrict__ biasbase, const float* __restrict__ resw1c,
    const float* __restrict__ qc0c, const float* __restrict__ qc1c,
    const float* __restrict__ ohrow, float* stg, int lane, v8f* d) {
  v8f h1[4];
#pragma unroll
  for (int j = 0; j < 4; ++j) {
    v8f acc = {};
    acc = wmma_f16(a0, bfrag_ld(fb, FRAG_W1(k, j, 0), lane), acc);
    acc = wmma_f16(a1, bfrag_ld(fb, FRAG_W1(k, j, 1), lane), acc);
    h1[j] = acc;
  }
  int n = lane & 15;
  int mb = (lane & 16) ? 8 : 0;
  const float* bb = biasbase + (k * B_ + b) * 64;
  const float* rw = resw1c + (k * B_ + b) * 64;
  const float* q0 = qc0c + k * 64;
  const float* q1 = qc1c + k * 64;
  const float* qp[8];
#pragma unroll
  for (int v = 0; v < 8; ++v) {
    int c = c0 + mb + v;
    qp[v] = (c == qtb && qtb < C_) ? rw : ((ohrow[c] != 0.0f) ? q1 : q0);
  }
#pragma unroll
  for (int j = 0; j < 4; ++j) {
    int col = (j << 4) + n;
    float base = bb[col];
#pragma unroll
    for (int v = 0; v < 8; ++v) {
      float x = h1[j][v] + base + qp[v][col];
      stg[(mb + v) * 64 + col] = x > 0.0f ? x : 0.0f;
    }
  }
  v16h ha0 = afrag_lds(stg, 0, lane);
  v16h ha1 = afrag_lds(stg, 1, lane);
  const float* b2 = fnb2 + k * 64;
#pragma unroll
  for (int j = 0; j < 4; ++j) {
    v8f acc = {};
    acc = wmma_f16(ha0, bfrag_ld(fb, FRAG_W2(k, j, 0), lane), acc);
    acc = wmma_f16(ha1, bfrag_ld(fb, FRAG_W2(k, j, 1), lane), acc);
    float bias = b2[(j << 4) + n];
#pragma unroll
    for (int v = 0; v < 8; ++v) {
      float x = acc[v] + bias;
      acc[v] = x > 0.0f ? x : 0.0f;
    }
    d[j] = acc;
  }
}

// -------------------------- small prep kernels -----------------------------
__global__ __launch_bounds__(256) void k_mean(const float* __restrict__ emb_x,
                                              float* __restrict__ sm) {
  int c = blockIdx.x * 256 + threadIdx.x;
  if (c < C_) {
    float s = 0.f;
#pragma unroll 8
    for (int e = 0; e < E_; ++e) s += emb_x[c * E_ + e];
    sm[c] = s * (1.0f / E_);
  }
}

__global__ __launch_bounds__(256) void k_se1(const float* __restrict__ sm,
                                             const float* __restrict__ w1,
                                             const float* __restrict__ b1,
                                             float* __restrict__ s1) {
  int j = blockIdx.x * 256 + threadIdx.x;
  if (j < 512) {
    float acc = b1[j];
    for (int c = 0; c < C_; ++c) acc += sm[c] * w1[c * 512 + j];
    s1[j] = acc > 0.f ? acc : 0.f;
  }
}

__global__ __launch_bounds__(256) void k_se2(const float* __restrict__ s1,
                                             const float* __restrict__ w2,
                                             const float* __restrict__ b2,
                                             float* __restrict__ s2) {
  int c = blockIdx.x * 256 + threadIdx.x;
  if (c < C_) {
    float acc = b2[c];
    for (int j = 0; j < 512; ++j) acc += s1[j] * w2[j * C_ + c];
    s2[c] = sigf(acc);
  }
}

__global__ __launch_bounds__(128) void k_prep_qc(const float* __restrict__ emb_c,
                                                 const float* __restrict__ fnW1,
                                                 float* __restrict__ qc0c,
                                                 float* __restrict__ qc1c) {
  int t = threadIdx.x;
  int k = t >> 6, j = t & 63;
  const float* W = fnW1 + k * 256 * 64;
  float a0 = 0.f, a1 = 0.f;
  for (int e = 0; e < 64; ++e) {
    float w = W[(192 + e) * 64 + j];
    a0 += emb_c[e] * w;
    a1 += emb_c[64 + e] * w;
  }
  qc0c[k * 64 + j] = a0;
  qc1c[k * 64 + j] = a1;
}

__global__ __launch_bounds__(64) void k_prep_b(
    const float* __restrict__ xt, const int* __restrict__ qt,
    const float* __restrict__ ht, const float* __restrict__ onehot,
    const float* __restrict__ emb_x, const float* __restrict__ emb_c,
    const float* __restrict__ s2, const float* __restrict__ fnW1,
    const float* __restrict__ fnb1, float* __restrict__ res_emb,
    float* __restrict__ selfht, float* __restrict__ biasbase,
    float* __restrict__ resw1c, float* __restrict__ invden,
    int* __restrict__ nzidx, int* __restrict__ nzcnt) {
  __shared__ float resh[64];
  __shared__ float sh[128];
  int b = blockIdx.x, e = threadIdx.x;
  int qtb = qt[b];
  const float* ohrow = onehot + (size_t)qtb * C_;
  float acc = 0.f;
  for (int c = 0; c < C_; ++c) {
    float mk = ohrow[c];
    if (mk != 0.f) acc += mk * emb_x[c * E_ + e] * s2[c];
  }
  float re = xt[b] * acc;
  resh[e] = re;
  res_emb[b * 64 + e] = re;
  if (e == 0) {
    int cnt = 0;
    for (int c = 1; c < C_; ++c)
      if (ohrow[c] != 0.f) { if (cnt < 8) nzidx[b * 8 + cnt] = c; ++cnt; }
    nzcnt[b] = cnt < 8 ? cnt : 8;
    invden[b] = 1.0f / (float)cnt;
  }
  int qtc = qtb < C_ ? qtb : C_ - 1;
  float qce = (qtb < C_) ? re : emb_c[((ohrow[qtc] != 0.f) ? 1 : 0) * 64 + e];
  sh[e]      = ht[((size_t)b * C_ + qtc) * 64 + e];
  sh[64 + e] = qce;
  selfht[b * 128 + e]      = sh[e];
  selfht[b * 128 + 64 + e] = sh[64 + e];
  __syncthreads();
#pragma unroll
  for (int k = 0; k < K_; ++k) {
    const float* W = fnW1 + k * 256 * 64;
    float sacc = fnb1[k * 64 + e];
    for (int d = 0; d < 128; ++d) sacc += sh[d] * W[d * 64 + e];
    biasbase[(k * B_ + b) * 64 + e] = sacc;
    float racc = 0.f;
    for (int d = 0; d < 64; ++d) racc += resh[d] * W[(192 + d) * 64 + e];
    resw1c[(k * B_ + b) * 64 + e] = racc;
  }
}

__global__ __launch_bounds__(64) void k_selffeat(
    const float* __restrict__ selfht, const float* __restrict__ W1,
    const float* __restrict__ b1, const float* __restrict__ W2,
    const float* __restrict__ b2, const float* __restrict__ g,
    const float* __restrict__ bt, float* __restrict__ selffeat) {
  __shared__ float h1[64 * 64];
  __shared__ float h2[64 * 64];
  int t = threadIdx.x;
  for (int j = 0; j < 64; ++j) {
    float acc = b1[j];
    for (int d = 0; d < 128; ++d) acc += selfht[t * 128 + d] * W1[d * 64 + j];
    h1[t * 64 + j] = acc > 0.f ? acc : 0.f;
  }
  __syncthreads();
  for (int j = 0; j < 64; ++j) {
    float acc = b2[j];
    for (int d = 0; d < 64; ++d) acc += h1[t * 64 + d] * W2[d * 64 + j];
    h2[t * 64 + j] = acc > 0.f ? acc : 0.f;
  }
  __syncthreads();
  float mu = 0.f, vq = 0.f;
  for (int r = 0; r < 64; ++r) { float x = h2[r * 64 + t]; mu += x; vq += x * x; }
  mu *= (1.0f / 64.0f);
  vq = vq * (1.0f / 64.0f) - mu * mu;
  float inv = rsqrtf(vq + 1e-5f);
  for (int r = 0; r < 64; ++r)
    selffeat[r * 64 + t] = (h2[r * 64 + t] - mu) * inv * g[t] + bt[t];
}

__global__ __launch_bounds__(256) void k_adj(const float* __restrict__ graphs,
                                             const int* __restrict__ nzidx,
                                             const int* __restrict__ nzcnt,
                                             const float* __restrict__ invden,
                                             float* __restrict__ adj) {
  int kk = blockIdx.x >> 6, b = blockIdx.x & 63;
  int bp = B_ - 1 - b;
  int cnt = nzcnt[bp];
  float inv = invden[bp];
  for (int c = threadIdx.x; c < C_; c += 256) {
    float s = 0.f;
    for (int i = 0; i < cnt; ++i)
      s += graphs[(size_t)kk * C_ * C_ + (size_t)nzidx[bp * 8 + i] * C_ + c];
    adj[((size_t)(kk * B_ + b)) * C_ + c] = s * inv;
  }
}

__global__ __launch_bounds__(256) void k_zero(float* __restrict__ p, int n) {
  int i = blockIdx.x * 256 + threadIdx.x;
  if (i < n) p[i] = 0.f;
}

// -------------------------- pass 1: fn batch-norm stats --------------------
__global__ __launch_bounds__(128) void k_stats(
    const float* __restrict__ ht, const float* __restrict__ onehot,
    const int* __restrict__ qt, const _Float16* __restrict__ fb,
    const float* __restrict__ fnb2, const float* __restrict__ biasbase,
    const float* __restrict__ resw1c, const float* __restrict__ qc0c,
    const float* __restrict__ qc1c, float* __restrict__ ssum,
    float* __restrict__ ssq) {
  __shared__ __align__(16) float htb[4][2][1024];
  __shared__ __align__(16) float stg[4][1024];
  int lane = threadIdx.x & 31;
  int w = threadIdx.x >> 5;
  int gw = (blockIdx.x << 2) + w;
  int nw = gridDim.x << 2;
  float accs[K_][4] = {};
  float accq[K_][4] = {};
#if USE_TDM
  int buf = 0;
  if (gw < NT_)
    tdm_load_tile(ht + ((size_t)((gw >> 7) * C_ + ((gw & 127) << 4))) * 64,
                  htb[w][0]);
#endif
  for (int t = gw; t < NT_; t += nw) {
    int b = t >> 7;
    int c0 = (t & 127) << 4;
#if USE_TDM
    int nxt = t + nw;
    if (nxt < NT_) {
      tdm_load_tile(ht + ((size_t)((nxt >> 7) * C_ + ((nxt & 127) << 4))) * 64,
                    htb[w][buf ^ 1]);
      __builtin_amdgcn_s_wait_tensorcnt((short)1);
    } else {
      __builtin_amdgcn_s_wait_tensorcnt((short)0);
    }
    const float* htw = htb[w][buf];
    buf ^= 1;
#else
    const float4* src = (const float4*)(ht + ((size_t)(b * C_ + c0)) * 64);
    float4* dst = (float4*)htb[w][0];
#pragma unroll
    for (int i = 0; i < 8; ++i) dst[lane + (i << 5)] = src[lane + (i << 5)];
    const float* htw = htb[w][0];
#endif
    v16h a0 = afrag_lds(htw, 0, lane);
    v16h a1 = afrag_lds(htw, 1, lane);
    int qtb = qt[b];
    const float* ohrow = onehot + (size_t)qtb * C_;
#pragma unroll
    for (int k = 0; k < K_; ++k) {
      v8f d[4];
      compute_h2(k, b, c0, qtb, a0, a1, fb, fnb2, biasbase, resw1c, qc0c, qc1c,
                 ohrow, stg[w], lane, d);
#pragma unroll
      for (int j = 0; j < 4; ++j) {
        float s = 0.f, q = 0.f;
#pragma unroll
        for (int v = 0; v < 8; ++v) { float x = d[j][v]; s += x; q += x * x; }
        accs[k][j] += s;
        accq[k][j] += q;
      }
    }
  }
#pragma unroll
  for (int k = 0; k < K_; ++k)
#pragma unroll
    for (int j = 0; j < 4; ++j) {
      float s = accs[k][j] + __shfl_xor(accs[k][j], 16, 32);
      float q = accq[k][j] + __shfl_xor(accq[k][j], 16, 32);
      if (lane < 16) {
        atomicAdd(&ssum[k * 64 + (j << 4) + lane], s);
        atomicAdd(&ssq[k * 64 + (j << 4) + lane], q);
      }
    }
}

__global__ __launch_bounds__(128) void k_finstats(
    const float* __restrict__ ssum, const float* __restrict__ ssq,
    const float* __restrict__ g, const float* __restrict__ bt,
    float* __restrict__ anorm, float* __restrict__ bnorm) {
  int t = threadIdx.x;
  const float invR = 1.0f / (float)(B_ * C_);
  float mu = ssum[t] * invR;
  float var = ssq[t] * invR - mu * mu;
  float a = g[t] * rsqrtf(var + 1e-5f);
  anorm[t] = a;
  bnorm[t] = bt[t] - mu * a;
}

// -------- pass 2: recompute fn, combine with adj, EAG gate, GRU, predict ---
__global__ __launch_bounds__(128) void k_pass2(
    const float* __restrict__ ht, const float* __restrict__ onehot,
    const int* __restrict__ qt, const _Float16* __restrict__ fb,
    const float* __restrict__ fnb2, const float* __restrict__ biasbase,
    const float* __restrict__ resw1c, const float* __restrict__ qc0c,
    const float* __restrict__ qc1c, const float* __restrict__ anorm,
    const float* __restrict__ bnorm, const float* __restrict__ adj,
    const float* __restrict__ selffeat, const float* __restrict__ eag_w,
    const float* __restrict__ ebv, const float* __restrict__ abv,
    const float* __restrict__ bih, const float* __restrict__ bhh,
    const float* __restrict__ pw, const float* __restrict__ pbp,
    float* __restrict__ y) {
  __shared__ __align__(16) float htb[4][2][1024];
  __shared__ __align__(16) float stg[4][1024];
  int lane = threadIdx.x & 31;
  int w = threadIdx.x >> 5;
  int gw = (blockIdx.x << 2) + w;
  int nw = gridDim.x << 2;
  int n = lane & 15;
  int mb = (lane & 16) ? 8 : 0;
  float pb = pbp[0];
#if USE_TDM
  int buf = 0;
  if (gw < NT_)
    tdm_load_tile(ht + ((size_t)((gw >> 7) * C_ + ((gw & 127) << 4))) * 64,
                  htb[w][0]);
#endif
  for (int t = gw; t < NT_; t += nw) {
    int b = t >> 7;
    int c0 = (t & 127) << 4;
#if USE_TDM
    int nxt = t + nw;
    if (nxt < NT_) {
      tdm_load_tile(ht + ((size_t)((nxt >> 7) * C_ + ((nxt & 127) << 4))) * 64,
                    htb[w][buf ^ 1]);
      __builtin_prefetch(adj + (size_t)(nxt >> 7) * C_ + ((nxt & 127) << 4), 0, 0);
      __builtin_amdgcn_s_wait_tensorcnt((short)1);
    } else {
      __builtin_amdgcn_s_wait_tensorcnt((short)0);
    }
    const float* htw = htb[w][buf];
    buf ^= 1;
#else
    const float4* src = (const float4*)(ht + ((size_t)(b * C_ + c0)) * 64);
    float4* dst = (float4*)htb[w][0];
#pragma unroll
    for (int i = 0; i < 8; ++i) dst[lane + (i << 5)] = src[lane + (i << 5)];
    const float* htw = htb[w][0];
#endif
    v16h a0 = afrag_lds(htw, 0, lane);
    v16h a1 = afrag_lds(htw, 1, lane);
    int qtb = qt[b];
    const float* ohrow = onehot + (size_t)qtb * C_;

    // ---- m_next tile in D layout ----
    v8f m[4] = {{}, {}, {}, {}};
#pragma unroll
    for (int k = 0; k < K_; ++k) {
      v8f d[4];
      compute_h2(k, b, c0, qtb, a0, a1, fb, fnb2, biasbase, resw1c, qc0c, qc1c,
                 ohrow, stg[w], lane, d);
      float av[8];
#pragma unroll
      for (int v = 0; v < 8; ++v)
        av[v] = adj[((size_t)(k * B_ + b)) * C_ + c0 + mb + v];
#pragma unroll
      for (int j = 0; j < 4; ++j) {
        int col = (j << 4) + n;
        float an = anorm[k * 64 + col], bn = bnorm[k * 64 + col];
#pragma unroll
        for (int v = 0; v < 8; ++v) m[j][v] += av[v] * (d[j][v] * an + bn);
      }
    }
    // override row qt[b] with self_feat (scatter dropped when qt>=C)
#pragma unroll
    for (int j = 0; j < 4; ++j) {
      float sf = selffeat[b * 64 + (j << 4) + n];
#pragma unroll
      for (int v = 0; v < 8; ++v) {
        bool sel = (qtb < C_) && ((c0 + mb + v) == qtb);
        m[j][v] = sel ? sf : m[j][v];
      }
    }
    // ---- EAG gate: eg = sigmoid(m@eW+eb), ag = tanh(m@aW+ab) ----
#pragma unroll
    for (int j = 0; j < 4; ++j)
#pragma unroll
      for (int v = 0; v < 8; ++v) stg[w][(mb + v) * 64 + (j << 4) + n] = m[j][v];
    v16h ma0 = afrag_lds(stg[w], 0, lane);
    v16h ma1 = afrag_lds(stg[w], 1, lane);
    float wc[8];
#pragma unroll
    for (int v = 0; v < 8; ++v) wc[v] = eag_w[c0 + mb + v];
#pragma unroll
    for (int j = 0; j < 4; ++j) {
      v8f eA = {};
      eA = wmma_f16(ma0, bfrag_ld(fb, FRAG_EW(j, 0), lane), eA);
      eA = wmma_f16(ma1, bfrag_ld(fb, FRAG_EW(j, 1), lane), eA);
      v8f aA = {};
      aA = wmma_f16(ma0, bfrag_ld(fb, FRAG_AW(j, 0), lane), aA);
      aA = wmma_f16(ma1, bfrag_ld(fb, FRAG_AW(j, 1), lane), aA);
      int col = (j << 4) + n;
      float eb_ = ebv[col], ab_ = abv[col];
#pragma unroll
      for (int v = 0; v < 8; ++v) {
        float eg = sigf(eA[v] + eb_);
        float ag = tanhf(aA[v] + ab_);
        float mm = m[j][v];
        float m2 = mm - wc[v] * eg * mm + wc[v] * ag;
        stg[w][(mb + v) * 64 + col] = m2;  // stage m2 for GRU input GEMM
      }
    }
    v16h g0 = afrag_lds(stg[w], 0, lane);
    v16h g1 = afrag_lds(stg[w], 1, lane);

    // ---- GRU: gi = m2@wih.T+bih, gh = ht@whh.T+bhh; fused gates + y dot ---
    float ydot[8] = {};
#pragma unroll
    for (int jt = 0; jt < 4; ++jt) {
      v8f gir = {}, giz = {}, gin = {}, ghr = {}, ghz = {}, ghn = {};
      gir = wmma_f16(g0, bfrag_ld(fb, FRAG_WIH(jt, 0), lane), gir);
      gir = wmma_f16(g1, bfrag_ld(fb, FRAG_WIH(jt, 1), lane), gir);
      giz = wmma_f16(g0, bfrag_ld(fb, FRAG_WIH(jt + 4, 0), lane), giz);
      giz = wmma_f16(g1, bfrag_ld(fb, FRAG_WIH(jt + 4, 1), lane), giz);
      gin = wmma_f16(g0, bfrag_ld(fb, FRAG_WIH(jt + 8, 0), lane), gin);
      gin = wmma_f16(g1, bfrag_ld(fb, FRAG_WIH(jt + 8, 1), lane), gin);
      ghr = wmma_f16(a0, bfrag_ld(fb, FRAG_WHH(jt, 0), lane), ghr);
      ghr = wmma_f16(a1, bfrag_ld(fb, FRAG_WHH(jt, 1), lane), ghr);
      ghz = wmma_f16(a0, bfrag_ld(fb, FRAG_WHH(jt + 4, 0), lane), ghz);
      ghz = wmma_f16(a1, bfrag_ld(fb, FRAG_WHH(jt + 4, 1), lane), ghz);
      ghn = wmma_f16(a0, bfrag_ld(fb, FRAG_WHH(jt + 8, 0), lane), ghn);
      ghn = wmma_f16(a1, bfrag_ld(fb, FRAG_WHH(jt + 8, 1), lane), ghn);
      int col = (jt << 4) + n;
      float bir = bih[col], biz = bih[64 + col], bin = bih[128 + col];
      float bhr = bhh[col], bhz = bhh[64 + col], bhn = bhh[128 + col];
      float pwv = pw[col];
#pragma unroll
      for (int v = 0; v < 8; ++v) {
        float r = sigf(gir[v] + bir + ghr[v] + bhr);
        float z = sigf(giz[v] + biz + ghz[v] + bhz);
        float nn = tanhf(gin[v] + bin + r * (ghn[v] + bhn));
        float hprev = htw[(mb + v) * 64 + col];
        float hn = (1.0f - z) * nn + z * hprev;
        ydot[v] += hn * pwv;
      }
    }
#pragma unroll
    for (int v = 0; v < 8; ++v) {
      float s = ydot[v];
      s += __shfl_xor(s, 1, 32);
      s += __shfl_xor(s, 2, 32);
      s += __shfl_xor(s, 4, 32);
      s += __shfl_xor(s, 8, 32);
      if ((lane & 15) == 0)
        y[(size_t)b * C_ + c0 + mb + v] = sigf(s + pb);
    }
  }
}

// ---------------------------------------------------------------------------
extern "C" void kernel_launch(void* const* d_in, const int* in_sizes, int n_in,
                              void* d_out, int out_size, void* d_ws, size_t ws_size,
                              hipStream_t stream) {
  (void)in_sizes; (void)n_in; (void)out_size; (void)ws_size;
  const float* xt     = (const float*)d_in[0];
  const int*   qt     = (const int*)d_in[1];
  const float* ht     = (const float*)d_in[2];
  const float* onehot = (const float*)d_in[3];
  const float* graphs = (const float*)d_in[4];
  const float* emb_x  = (const float*)d_in[5];
  const float* se_w1  = (const float*)d_in[6];
  const float* se_b1  = (const float*)d_in[7];
  const float* se_w2  = (const float*)d_in[8];
  const float* se_b2  = (const float*)d_in[9];
  const float* emb_c  = (const float*)d_in[10];
  const float* fsW1   = (const float*)d_in[11];
  const float* fsb1   = (const float*)d_in[12];
  const float* fsW2   = (const float*)d_in[13];
  const float* fsb2   = (const float*)d_in[14];
  const float* fsg    = (const float*)d_in[15];
  const float* fsbt   = (const float*)d_in[16];
  const float* fnW1   = (const float*)d_in[17];
  const float* fnb1   = (const float*)d_in[18];
  const float* fnW2   = (const float*)d_in[19];
  const float* fnb2   = (const float*)d_in[20];
  const float* fng    = (const float*)d_in[21];
  const float* fnbt   = (const float*)d_in[22];
  const float* eag_w  = (const float*)d_in[23];
  const float* eW     = (const float*)d_in[24];
  const float* ebv    = (const float*)d_in[25];
  const float* aW     = (const float*)d_in[26];
  const float* abv    = (const float*)d_in[27];
  const float* wih    = (const float*)d_in[28];
  const float* whh    = (const float*)d_in[29];
  const float* bih    = (const float*)d_in[30];
  const float* bhh    = (const float*)d_in[31];
  const float* pw     = (const float*)d_in[32];
  const float* pbp    = (const float*)d_in[33];
  float* y = (float*)d_out;

  // workspace carve: fragment buffer first (32B aligned), then f32 scratch
  _Float16* fragbuf = (_Float16*)d_ws;       // 96 frags * 512 halves = 96 KB
  float* fp       = (float*)d_ws + 24576;    // after 96 KB
  float* sm       = fp;               // 2048
  float* s1       = sm + 2048;        // 512
  float* s2       = s1 + 512;         // 2048
  float* res_emb  = s2 + 2048;        // 4096
  float* selfht   = res_emb + 4096;   // 8192
  float* selffeat = selfht + 8192;    // 4096
  float* biasbase = selffeat + 4096;  // 8192
  float* resw1c   = biasbase + 8192;  // 8192
  float* qc0c     = resw1c + 8192;    // 128
  float* qc1c     = qc0c + 128;       // 128
  float* invden   = qc1c + 128;       // 64
  float* ssum     = invden + 64;      // 128
  float* ssq      = ssum + 128;       // 128
  float* anorm    = ssq + 128;        // 128
  float* bnorm    = anorm + 128;      // 128
  float* adj      = bnorm + 128;      // K*B*C = 262144
  int*   nzidx    = (int*)(adj + (size_t)K_ * B_ * C_);  // 512
  int*   nzcnt    = nzidx + 512;      // 64

  k_packfrags<<<NFRAG, 32, 0, stream>>>(fnW1, fnW2, eW, aW, wih, whh, fragbuf);
  // squeeze-excite
  k_mean<<<8, 256, 0, stream>>>(emb_x, sm);
  k_se1<<<2, 256, 0, stream>>>(sm, se_w1, se_b1, s1);
  k_se2<<<8, 256, 0, stream>>>(s1, se_w2, se_b2, s2);
  // qc column-bias precompute and per-b precompute
  k_prep_qc<<<1, 128, 0, stream>>>(emb_c, fnW1, qc0c, qc1c);
  k_prep_b<<<B_, 64, 0, stream>>>(xt, qt, ht, onehot, emb_x, emb_c, s2, fnW1,
                                  fnb1, res_emb, selfht, biasbase, resw1c,
                                  invden, nzidx, nzcnt);
  k_selffeat<<<1, 64, 0, stream>>>(selfht, fsW1, fsb1, fsW2, fsb2, fsg, fsbt,
                                   selffeat);
  k_adj<<<K_ * B_, 256, 0, stream>>>(graphs, nzidx, nzcnt, invden, adj);
  // fn batch-norm statistics (pass 1, WMMA)
  k_zero<<<1, 256, 0, stream>>>(ssum, 256);
  k_stats<<<256, 128, 0, stream>>>(ht, onehot, qt, fragbuf, fnb2, biasbase,
                                   resw1c, qc0c, qc1c, ssum, ssq);
  k_finstats<<<1, 128, 0, stream>>>(ssum, ssq, fng, fnbt, anorm, bnorm);
  // fused recompute + combine + EAG + GRU + predict (pass 2, WMMA)
  k_pass2<<<512, 128, 0, stream>>>(ht, onehot, qt, fragbuf, fnb2, biasbase,
                                   resw1c, qc0c, qc1c, anorm, bnorm, adj,
                                   selffeat, eag_w, ebv, abv, bih, bhh, pw, pbp,
                                   y);
}
